// Baseline_Enc_Dec_model_24816321036876
// MI455X (gfx1250) — matrix operand order repaired and checked by reading, not compiled, not run on hardware
//
#include <hip/hip_runtime.h>

// ============================================================================
// Enc-Dec GRU model on MI455X (gfx1250), bf16 WMMA implementation.
//
// Shapes: B=1024, T=100, PRED=30, E=512, H=1024, IN=2.
// Dominant cost: per-step GEMMs (B x 3H x {E|H}) -> v_wmma_f32_16x16x32_bf16.
// Weights (bf16, ~42MB total) stay resident in the 192MB L2 across steps;
// activations ~6MB/step at 23.3TB/s -> per-step cost is WMMA throughput.
//
// Layout trick: C = X @ W^T means WMMA B-matrix column n == weight row n,
// so row-major bf16 weights load straight into B-register layout with two
// aligned 16B loads per lane; row-major activations load into A-layout with
// two aligned 16B loads per lane. No transpose, no LDS staging needed.
//
// Register blocking: each wave computes a 32(M) x 32(N) tile of h_new:
// 2 A frags x 6 B frags (3 gates x 2 N-subtiles) -> 12 WMMAs per 16 b128
// loads per 32-K step (~24 FLOP/byte from cache), 16 f32 accumulators.
//
// Workspace budget (~355 MB): bf16 weights 42MB + emb (B,T,E) 100MB +
// enc1 outputs (B,T,H) 200MB + ~13MB of hidden/scratch buffers.
// ============================================================================

#define BB   1024
#define TT   100
#define PRED 30
#define EE   512
#define HH   1024

typedef __bf16 bf16_t;
typedef __attribute__((ext_vector_type(16))) __bf16 v16bf;
typedef __attribute__((ext_vector_type(8)))  __bf16 v8bf;
typedef __attribute__((ext_vector_type(8)))  float  v8f;

union V16U { v16bf v; v8bf h[2]; };

// ---------------------------------------------------------------------------
// WMMA helpers
// ---------------------------------------------------------------------------
__device__ __forceinline__ v8f wmma_bf16(v16bf a, v16bf b, v8f c) {
  // emits v_wmma_f32_16x16x32_bf16
  return __builtin_amdgcn_wmma_f32_16x16x32_bf16(
      /*neg_a=*/false, a, /*neg_b=*/false, b,
      /*c_mod=*/(short)0, c, /*reuse_a=*/false, /*reuse_b=*/false);
}

// A-matrix 16x32 bf16 tile from row-major [rows x lda] activations.
// ISA layout: lanes 0-15 hold row m0+lane, K = {k..k+7} in v0..3 (pairs) and
// {k+16..k+23} in v4..7; lanes 16-31 hold K shifted by +8. Both halves are
// contiguous 16B chunks -> two global_load_b128 per lane, no shuffle.
__device__ __forceinline__ v16bf load_A(const bf16_t* __restrict__ A, int lda,
                                        int m0, int k, int lane) {
  const int row = m0 + (lane & 15);
  const int kb  = k + ((lane >> 4) << 3);
  const bf16_t* p = A + (size_t)row * (size_t)lda + kb;
  V16U u;
  u.h[0] = *(const v8bf*)(p);        // K = kb   .. kb+7
  u.h[1] = *(const v8bf*)(p + 16);   // K = kb+16.. kb+23
  return u.v;
}

// B-matrix 32x16 bf16 tile for C = X @ W^T with W row-major [3H x K]:
// B[k,n] = W[n0+n, k]. Lane n holds weight row n with the same K order as A
// (lanes 0-15: K=k..k+7,k+16..k+23; lanes 16-31: +8) -> two aligned 16B loads.
__device__ __forceinline__ v16bf load_B(const bf16_t* __restrict__ W, int K,
                                        int n0, int k, int lane) {
  const int row = n0 + (lane & 15);
  const int kb  = k + ((lane >> 4) << 3);
  const bf16_t* p = W + (size_t)row * (size_t)K + kb;
  V16U u;
  u.h[0] = *(const v8bf*)(p);
  u.h[1] = *(const v8bf*)(p + 16);
  return u.v;
}

__device__ __forceinline__ float sigmoidf_(float x) {
  return 1.0f / (1.0f + __expf(-x));
}

// ---------------------------------------------------------------------------
// Fused GRU step:  h_new = (1-z)*n + z*h,  r/z = sigmoid(gx+gh),
// n = tanh(gx_n + r*gh_n).  gx = X@Wih^T + bih, gh = H@Whh^T + bhh.
// One wave computes a 32x32 tile of h_new: 2 M-subtiles x 2 N-subtiles.
// Accumulators: {r,z,xn,hn} x 2 x 2 (xn/hn separate: gh_n is scaled by r
// before tanh). Each B fragment feeds 2 WMMAs, each A fragment feeds 6.
// Grid: (B/32)*(H/32)/8 = 128 blocks x 256 threads (8 waves).
// ---------------------------------------------------------------------------
__global__ __launch_bounds__(256) void gru_step_wmma(
    const bf16_t* __restrict__ X, int ldx, int Kx,
    const bf16_t* __restrict__ Hprev, int ldh,
    bf16_t* __restrict__ Hnew, int ldn,
    const bf16_t* __restrict__ Wih,   // (3H, Kx) row-major bf16
    const bf16_t* __restrict__ Whh,   // (3H, H)  row-major bf16
    const float* __restrict__ bih,
    const float* __restrict__ bhh) {
  const int lane = threadIdx.x & 31;
  const int wave = threadIdx.x >> 5;
  const int tile = blockIdx.x * 8 + wave;
  const int mTiles = BB / 32;                 // 32
  const int m0 = (tile % mTiles) * 32;        // batch tile (32 rows)
  const int n0 = (tile / mTiles) * 32;        // hidden-unit tile (32 cols)

  v8f acc_r[2][2]  = {{{}, {}}, {{}, {}}};
  v8f acc_z[2][2]  = {{{}, {}}, {{}, {}}};
  v8f acc_xn[2][2] = {{{}, {}}, {{}, {}}};
  v8f acc_hn[2][2] = {{{}, {}}, {{}, {}}};

  // ---- gx = X @ Wih^T (r, z, n gate columns at n0, n0+H, n0+2H) ----
  for (int k = 0; k < Kx; k += 32) {
    v16bf a[2], br[2], bz[2], bn[2];
#pragma unroll
    for (int mi = 0; mi < 2; ++mi) a[mi] = load_A(X, ldx, m0 + 16 * mi, k, lane);
#pragma unroll
    for (int ni = 0; ni < 2; ++ni) {
      const int nb = n0 + 16 * ni;
      br[ni] = load_B(Wih, Kx, nb,          k, lane);
      bz[ni] = load_B(Wih, Kx, nb + HH,     k, lane);
      bn[ni] = load_B(Wih, Kx, nb + 2 * HH, k, lane);
    }
#pragma unroll
    for (int mi = 0; mi < 2; ++mi)
#pragma unroll
      for (int ni = 0; ni < 2; ++ni) {
        acc_r[mi][ni]  = wmma_bf16(a[mi], br[ni], acc_r[mi][ni]);
        acc_z[mi][ni]  = wmma_bf16(a[mi], bz[ni], acc_z[mi][ni]);
        acc_xn[mi][ni] = wmma_bf16(a[mi], bn[ni], acc_xn[mi][ni]);
      }
  }

  // ---- gh = Hprev @ Whh^T ----
  for (int k = 0; k < HH; k += 32) {
    v16bf a[2], br[2], bz[2], bn[2];
#pragma unroll
    for (int mi = 0; mi < 2; ++mi) a[mi] = load_A(Hprev, ldh, m0 + 16 * mi, k, lane);
#pragma unroll
    for (int ni = 0; ni < 2; ++ni) {
      const int nb = n0 + 16 * ni;
      br[ni] = load_B(Whh, HH, nb,          k, lane);
      bz[ni] = load_B(Whh, HH, nb + HH,     k, lane);
      bn[ni] = load_B(Whh, HH, nb + 2 * HH, k, lane);
    }
#pragma unroll
    for (int mi = 0; mi < 2; ++mi)
#pragma unroll
      for (int ni = 0; ni < 2; ++ni) {
        acc_r[mi][ni]  = wmma_bf16(a[mi], br[ni], acc_r[mi][ni]);
        acc_z[mi][ni]  = wmma_bf16(a[mi], bz[ni], acc_z[mi][ni]);
        acc_hn[mi][ni] = wmma_bf16(a[mi], bn[ni], acc_hn[mi][ni]);
      }
  }

  // ---- epilogue: C layout -> VGPR i holds (m = msub + i + 8*(lane>=16),
  //                                          n = nsub + lane%16) ----
#pragma unroll
  for (int ni = 0; ni < 2; ++ni) {
    const int n    = n0 + 16 * ni + (lane & 15);
    const float br_b = bih[n]          + bhh[n];
    const float bz_b = bih[HH + n]     + bhh[HH + n];
    const float bxn  = bih[2 * HH + n];
    const float bhn  = bhh[2 * HH + n];
#pragma unroll
    for (int mi = 0; mi < 2; ++mi) {
      const int mbase = m0 + 16 * mi + ((lane >> 4) << 3);
#pragma unroll
      for (int i = 0; i < 8; ++i) {
        const int m = mbase + i;
        const float r  = sigmoidf_(acc_r[mi][ni][i] + br_b);
        const float z  = sigmoidf_(acc_z[mi][ni][i] + bz_b);
        const float nn = tanhf(acc_xn[mi][ni][i] + bxn +
                               r * (acc_hn[mi][ni][i] + bhn));
        const float hp = (float)Hprev[(size_t)m * (size_t)ldh + n];
        Hnew[(size_t)m * (size_t)ldn + n] = (bf16_t)((1.0f - z) * nn + z * hp);
      }
    }
  }
}

// ---------------------------------------------------------------------------
// Small support kernels
// ---------------------------------------------------------------------------
__global__ __launch_bounds__(256) void f32_to_bf16(const float* __restrict__ s,
                                                   bf16_t* __restrict__ d, int n) {
  for (int i = blockIdx.x * blockDim.x + threadIdx.x; i < n;
       i += gridDim.x * blockDim.x)
    d[i] = (bf16_t)s[i];
}

__global__ __launch_bounds__(256) void fill_zero_bf16(bf16_t* __restrict__ d, int n) {
  for (int i = blockIdx.x * blockDim.x + threadIdx.x; i < n;
       i += gridDim.x * blockDim.x)
    d[i] = (bf16_t)0.0f;
}

// emb[b,t,e] = tanh(obs[b,t,0]*We[e,0] + obs[b,t,1]*We[e,1] + be[e])  (K=2)
__global__ __launch_bounds__(256) void embed_all(
    const float* __restrict__ obs, const bf16_t* __restrict__ We,
    const float* __restrict__ be, bf16_t* __restrict__ emb, int total) {
  for (int idx = blockIdx.x * blockDim.x + threadIdx.x; idx < total;
       idx += gridDim.x * blockDim.x) {
    const int e  = idx % EE;
    const int bt = idx / EE;
    const float x0 = obs[(size_t)bt * 2 + 0];
    const float x1 = obs[(size_t)bt * 2 + 1];
    const float v  = tanhf(x0 * (float)We[e * 2 + 0] +
                           x1 * (float)We[e * 2 + 1] + be[e]);
    emb[idx] = (bf16_t)v;
  }
}

// demb[b,e] = tanh(last[b,0]*Wed[e,0] + last[b,1]*Wed[e,1] + bed[e])
__global__ __launch_bounds__(256) void dec_embed(
    const float* __restrict__ last, const bf16_t* __restrict__ Wed,
    const float* __restrict__ bed, bf16_t* __restrict__ demb) {
  const int total = BB * EE;
  for (int idx = blockIdx.x * blockDim.x + threadIdx.x; idx < total;
       idx += gridDim.x * blockDim.x) {
    const int e = idx % EE;
    const int b = idx / EE;
    const float v = tanhf(last[b * 2 + 0] * (float)Wed[e * 2 + 0] +
                          last[b * 2 + 1] * (float)Wed[e * 2 + 1] + bed[e]);
    demb[idx] = (bf16_t)v;
  }
}

// out[b,c] = h[b,:] . Wout[c,:] + bout[c];  writes fp32 to d_out slot (if any)
// and to `last` (decoder feedback). N=2, so plain vectorized dot per thread.
__global__ __launch_bounds__(256) void out_proj(
    const bf16_t* __restrict__ Hin, int ldh,
    const bf16_t* __restrict__ Wout, const float* __restrict__ bout,
    float* __restrict__ out_dst, int out_stride,   // out_dst may be nullptr
    float* __restrict__ last) {
  const int idx = blockIdx.x * blockDim.x + threadIdx.x;
  if (idx >= BB * 2) return;
  const int b = idx >> 1;
  const int c = idx & 1;
  const bf16_t* h = Hin + (size_t)b * (size_t)ldh;
  const bf16_t* w = Wout + (size_t)c * HH;
  float acc = bout[c];
#pragma unroll 4
  for (int k = 0; k < HH; k += 8) {
    const v8bf hv = *(const v8bf*)(h + k);
    const v8bf wv = *(const v8bf*)(w + k);
#pragma unroll
    for (int j = 0; j < 8; ++j) acc += (float)hv[j] * (float)wv[j];
  }
  if (out_dst) out_dst[(size_t)b * out_stride + c] = acc;
  last[idx] = acc;
}

// ---------------------------------------------------------------------------
// Host-side orchestration
// ---------------------------------------------------------------------------
static inline size_t align256(size_t x) { return (x + 255) & ~(size_t)255; }

extern "C" void kernel_launch(void* const* d_in, const int* in_sizes, int n_in,
                              void* d_out, int out_size, void* d_ws, size_t ws_size,
                              hipStream_t stream) {
  (void)in_sizes; (void)n_in; (void)out_size; (void)ws_size;

  // Inputs (setup_inputs order)
  const float* obs      = (const float*)d_in[0];
  const float* We_f     = (const float*)d_in[1];
  const float* be       = (const float*)d_in[2];
  const float* Wed_f    = (const float*)d_in[3];
  const float* bed      = (const float*)d_in[4];
  const float* e1Wih_f  = (const float*)d_in[5];
  const float* e1Whh_f  = (const float*)d_in[6];
  const float* e1bih    = (const float*)d_in[7];
  const float* e1bhh    = (const float*)d_in[8];
  const float* e2Wih_f  = (const float*)d_in[9];
  const float* e2Whh_f  = (const float*)d_in[10];
  const float* e2bih    = (const float*)d_in[11];
  const float* e2bhh    = (const float*)d_in[12];
  const float* d1Wih_f  = (const float*)d_in[13];
  const float* d1Whh_f  = (const float*)d_in[14];
  const float* d1bih    = (const float*)d_in[15];
  const float* d1bhh    = (const float*)d_in[16];
  const float* d2Wih_f  = (const float*)d_in[17];
  const float* d2Whh_f  = (const float*)d_in[18];
  const float* d2bih    = (const float*)d_in[19];
  const float* d2bhh    = (const float*)d_in[20];
  const float* Wout_f   = (const float*)d_in[21];
  const float* bout     = (const float*)d_in[22];
  float* out = (float*)d_out;   // (B, PRED, 2) fp32

  // ---- workspace carve-up ----
  char* w = (char*)d_ws;
  size_t off = 0;
  auto abf = [&](size_t elems) -> bf16_t* {
    bf16_t* p = (bf16_t*)(w + off);
    off = align256(off + elems * sizeof(bf16_t));
    return p;
  };
  auto af32 = [&](size_t elems) -> float* {
    float* p = (float*)(w + off);
    off = align256(off + elems * sizeof(float));
    return p;
  };

  const size_t szIhE = (size_t)3 * HH * EE;   // (3H,E)
  const size_t szIhH = (size_t)3 * HH * HH;   // (3H,H)

  bf16_t* wWe   = abf((size_t)EE * 2);
  bf16_t* wWed  = abf((size_t)EE * 2);
  bf16_t* wWout = abf((size_t)2 * HH);
  bf16_t* wE1ih = abf(szIhE);
  bf16_t* wE1hh = abf(szIhH);
  bf16_t* wE2ih = abf(szIhH);
  bf16_t* wE2hh = abf(szIhH);
  bf16_t* wD1ih = abf(szIhE);
  bf16_t* wD1hh = abf(szIhH);
  bf16_t* wD2ih = abf(szIhH);
  bf16_t* wD2hh = abf(szIhH);
  bf16_t* emb   = abf((size_t)BB * TT * EE);  // (B,T,E) bf16, ~100MB
  bf16_t* e1out = abf((size_t)BB * TT * HH);  // (B,T,H) bf16, ~200MB
  bf16_t* zeros = abf((size_t)BB * HH);
  bf16_t* h2a   = abf((size_t)BB * HH);
  bf16_t* h2b   = abf((size_t)BB * HH);
  bf16_t* dX    = abf((size_t)BB * HH);
  bf16_t* dY    = abf((size_t)BB * HH);
  bf16_t* dZ    = abf((size_t)BB * HH);
  bf16_t* demb  = abf((size_t)BB * EE);
  float*  last  = af32((size_t)BB * 2);

  // ---- weight conversion fp32 -> bf16 (stream-ordered, capture-safe) ----
  auto cvt = [&](const float* s, bf16_t* d, size_t n) {
    int blocks = (int)((n + 255) / 256);
    if (blocks > 2048) blocks = 2048;
    f32_to_bf16<<<blocks, 256, 0, stream>>>(s, d, (int)n);
  };
  cvt(We_f,    wWe,   (size_t)EE * 2);
  cvt(Wed_f,   wWed,  (size_t)EE * 2);
  cvt(Wout_f,  wWout, (size_t)2 * HH);
  cvt(e1Wih_f, wE1ih, szIhE);
  cvt(e1Whh_f, wE1hh, szIhH);
  cvt(e2Wih_f, wE2ih, szIhH);
  cvt(e2Whh_f, wE2hh, szIhH);
  cvt(d1Wih_f, wD1ih, szIhE);
  cvt(d1Whh_f, wD1hh, szIhH);
  cvt(d2Wih_f, wD2ih, szIhH);
  cvt(d2Whh_f, wD2hh, szIhH);

  fill_zero_bf16<<<1024, 256, 0, stream>>>(zeros, BB * HH);

  // ---- encoder embedding for all T (one big elementwise pass) ----
  embed_all<<<4096, 256, 0, stream>>>(obs, wWe, be, emb, BB * TT * EE);

  const int gruBlocks = (BB / 32) * (HH / 32) / 8;   // 128 blocks x 8 waves

  // ---- encoder layer 1: h0 = 0; outputs stored (B,T,H) for layer 2 ----
  for (int t = 0; t < TT; ++t) {
    const bf16_t* X  = emb + (size_t)t * EE;           // rows strided by T*E
    const bf16_t* Hp = (t == 0) ? zeros : (e1out + (size_t)(t - 1) * HH);
    const int ldh    = (t == 0) ? HH : TT * HH;
    gru_step_wmma<<<gruBlocks, 256, 0, stream>>>(
        X, TT * EE, EE, Hp, ldh, e1out + (size_t)t * HH, TT * HH,
        wE1ih, wE1hh, e1bih, e1bhh);
  }

  // ---- encoder layer 2: quirk -> h0 = enc1 FINAL hidden ----
  bf16_t* ping[2] = {h2a, h2b};
  for (int t = 0; t < TT; ++t) {
    const bf16_t* X  = e1out + (size_t)t * HH;
    const bf16_t* Hp = (t == 0) ? (e1out + (size_t)(TT - 1) * HH)
                                : ping[(t - 1) & 1];
    const int ldh    = (t == 0) ? TT * HH : HH;
    gru_step_wmma<<<gruBlocks, 256, 0, stream>>>(
        X, TT * HH, HH, Hp, ldh, ping[t & 1], HH,
        wE2ih, wE2hh, e2bih, e2bhh);
    if (t == TT - 2) {
      // out_loc0 = enc2[:, -2, :] @ Wout^T + bout  -> initial `last`
      out_proj<<<8, 256, 0, stream>>>(ping[t & 1], HH, wWout, bout,
                                      nullptr, 0, last);
    }
  }

  // ---- autoregressive decoder: shared hidden, 30 steps ----
  bf16_t* P = ping[(TT - 1) & 1];                  // enc2 final hidden
  bf16_t* tgt[2] = {dX, dZ};                       // alternate, never == P
  for (int s = 0; s < PRED; ++s) {
    bf16_t* h2 = tgt[s & 1];
    dec_embed<<<2048, 256, 0, stream>>>(last, wWed, bed, demb);
    // h1 = GRUCell(dec1; x=demb, h=P)
    gru_step_wmma<<<gruBlocks, 256, 0, stream>>>(
        demb, EE, EE, P, HH, dY, HH, wD1ih, wD1hh, d1bih, d1bhh);
    // h2 = GRUCell(dec2; x=h1, h=h1)   (source quirk preserved)
    gru_step_wmma<<<gruBlocks, 256, 0, stream>>>(
        dY, HH, HH, dY, HH, h2, HH, wD2ih, wD2hh, d2bih, d2bhh);
    // preds[:, s, :] = h2 @ Wout^T + bout; also feedback into `last`
    out_proj<<<8, 256, 0, stream>>>(h2, HH, wWout, bout,
                                    out + s * 2, PRED * 2, last);
    P = h2;
  }
}
